// SpiralEnblock_45810121179171
// MI455X (gfx1250) — matrix operand order repaired and checked by reading, not compiled, not run on hardware
//
#include <hip/hip_runtime.h>

typedef __bf16 v16bf __attribute__((ext_vector_type(16)));
typedef float  v8f   __attribute__((ext_vector_type(8)));

#define B_    8
#define N_    50000
#define CIN   32
#define S_    9
#define COUT  64
#define M_    12500
#define NNZ_  50000

static __device__ __forceinline__ unsigned int pack_bf16(float lo, float hi) {
    __bf16 l = (__bf16)lo;
    __bf16 h = (__bf16)hi;
    unsigned short ul = __builtin_bit_cast(unsigned short, l);
    unsigned short uh = __builtin_bit_cast(unsigned short, h);
    return (unsigned int)ul | ((unsigned int)uh << 16);
}

// Branchless ELU(alpha=1): max(h,0) + (exp(min(h,0)) - 1).
// For h>0 the exp term is exactly 0; for h<=0 the max term is 0.
// Lowers to v_min/v_max/v_exp_f32/v_add -- no EXEC divergence, no libm poly.
static __device__ __forceinline__ float elu_fast(float h) {
    return fmaxf(h, 0.0f) + (__expf(fminf(h, 0.0f)) - 1.0f);
}

__global__ void zero_out_kernel(float* __restrict__ out, int n) {
    int i = blockIdx.x * blockDim.x + threadIdx.x;
    if (i < n) out[i] = 0.0f;
}

// One block = 128 threads = 4 waves. Each wave fuses 2 edges:
//   rows 0..7  = batches of edge e0, rows 8..15 = batches of edge e1.
// Per wave: 9 K-steps (one per spiral index) x 4 N-tiles of
// v_wmma_f32_16x16x32_bf16, then bias + ELU + value-scale + atomic scatter.
__global__ __launch_bounds__(128) void spiral_fused_kernel(
    const float* __restrict__ x,      // (B, N, CIN)
    const float* __restrict__ W,      // (S*CIN, COUT)
    const float* __restrict__ bias,   // (COUT,)
    const float* __restrict__ value,  // (NNZ,)
    const int*   __restrict__ spir,   // (N, S)
    const int*   __restrict__ rowi,   // (NNZ,)
    const int*   __restrict__ coli,   // (NNZ,)
    float*       __restrict__ out)    // (B, M, COUT)
{
    // W pre-packed into WMMA B-fragment layout (bf16 pairs):
    // index = ((s*4 + nt)*2 + khalf)*128 + n*8 + i
    // value = pack(W[s*32+khalf*8+2i+(i&4)*2][nt*16+n], W[s*32+khalf*8+2i+(i&4)*2+1][nt*16+n])
    __shared__ __align__(32) unsigned int wlds[S_ * 4 * 2 * 16 * 8]; // 9216 u32 = 36 KB

    const int tid = threadIdx.x;
    for (int t = tid; t < S_ * 4 * 2 * 16 * 8; t += 128) {
        int i  = t & 7;
        int n  = (t >> 3) & 15;
        int kh = (t >> 7) & 1;
        int nt = (t >> 8) & 3;
        int s  = t >> 10;
        int k0 = s * 32 + kh * 8 + 2 * i + (i & 4) * 2;
        int c  = nt * 16 + n;
        wlds[t] = pack_bf16(W[(size_t)k0 * COUT + c], W[(size_t)(k0 + 1) * COUT + c]);
    }
    __syncthreads();

    const int lane  = tid & 31;
    const int wave  = tid >> 5;
    const int eBase = blockIdx.x * 8 + wave * 2;

    const int mrow = lane & 15;   // WMMA row (A) / output column-in-tile (C)
    const int half = lane >> 4;   // K-half for A/B fragments; edge select for C

    // ---- A-side addressing: row mrow -> (edge, batch) ----
    const int eA = eBase + (mrow >> 3);
    const int bA = mrow & 7;
    const int nA = coli[eA];
    const float* xb = x + (size_t)bA * ((size_t)N_ * CIN);

    v8f acc[4];
    #pragma unroll
    for (int nt = 0; nt < 4; ++nt) acc[nt] = (v8f){};

    for (int s = 0; s < S_; ++s) {
        const int j = spir[nA * S_ + s];
        const float4* p4 = (const float4*)(xb + (size_t)j * CIN + half * 8);
        // lane loads K = lo..lo+7 and K = lo+16..lo+23 (lo = half*8)
        const float4 u0 = p4[0];
        const float4 u1 = p4[1];
        const float4 u2 = p4[4];
        const float4 u3 = p4[5];
        const float a0[8] = {u0.x, u0.y, u0.z, u0.w, u1.x, u1.y, u1.z, u1.w};
        const float a1[8] = {u2.x, u2.y, u2.z, u2.w, u3.x, u3.y, u3.z, u3.w};
        v16bf A;
        #pragma unroll
        for (int q = 0; q < 8; ++q) {
            A[q]     = (__bf16)a0[q];
            A[8 + q] = (__bf16)a1[q];
        }

        #pragma unroll
        for (int nt = 0; nt < 4; ++nt) {
            const unsigned int* bp =
                &wlds[(((s * 4 + nt) * 2 + half) * 16 + mrow) * 8];
            v16bf Bf = *(const v16bf*)bp;   // two ds_load_b128
            acc[nt] = __builtin_amdgcn_wmma_f32_16x16x32_bf16(
                false, A, false, Bf, (short)0, acc[nt], false, false);
        }
    }

    // ---- Epilogue: C/D layout -> lanes<16 = edge0 rows 0..7 (b = vgpr idx),
    //                              lanes>=16 = edge1 rows 8..15 ----
    const int   eC  = eBase + half;
    const float val = value[eC];
    const int   r   = rowi[eC];
    float* outc = out + (size_t)r * COUT;

    #pragma unroll
    for (int nt = 0; nt < 4; ++nt) {
        const int   ocol = nt * 16 + mrow;
        const float bs   = bias[ocol];
        #pragma unroll
        for (int i = 0; i < 8; ++i) {
            const float h = elu_fast(acc[nt][i] + bs);
            atomicAdd(outc + (size_t)i * ((size_t)M_ * COUT) + ocol, h * val);
        }
    }
}

extern "C" void kernel_launch(void* const* d_in, const int* in_sizes, int n_in,
                              void* d_out, int out_size, void* d_ws, size_t ws_size,
                              hipStream_t stream) {
    (void)in_sizes; (void)n_in; (void)d_ws; (void)ws_size; (void)out_size;
    const float* x     = (const float*)d_in[0];
    const float* W     = (const float*)d_in[1];
    const float* bias  = (const float*)d_in[2];
    const float* value = (const float*)d_in[3];
    const int*   spir  = (const int*)d_in[4];
    const int*   rowi  = (const int*)d_in[5];
    const int*   coli  = (const int*)d_in[6];
    float*       out   = (float*)d_out;

    const int total = B_ * M_ * COUT;  // 6.4M floats
    zero_out_kernel<<<(total + 255) / 256, 256, 0, stream>>>(out, total);

    // 2 edges/wave * 4 waves/block = 8 edges/block; NNZ = 50000 = 6250 * 8
    spiral_fused_kernel<<<NNZ_ / 8, 128, 0, stream>>>(
        x, W, bias, value, spir, rowi, coli, out);
}
